// GroupedQueryAttention_74028056314447
// MI455X (gfx1250) — matrix operand order repaired and checked by reading, not compiled, not run on hardware
//
#include <hip/hip_runtime.h>

// GQA forward for MI455X (gfx1250, wave32, WMMA + async/TDM tile staging).
// B=2, S=2048, H=2048, NH=16, NKV=4, HD=128, N_REP=4.

#define B_ 2
#define S_ 2048
#define H_ 2048
#define NH_ 16
#define NKV_ 4
#define HD_ 128
#define SCALE_ 0.08838834764831845f

typedef __attribute__((ext_vector_type(16))) __bf16 v16bf;
typedef __attribute__((ext_vector_type(8)))  float v8f;
typedef __attribute__((ext_vector_type(4)))  unsigned int v4u;
typedef __attribute__((ext_vector_type(8)))  int v8i;
typedef __attribute__((ext_vector_type(4)))  int v4i;

union FragBf16 { v16bf v; uint4 u[2]; };

static __device__ inline unsigned short f2bf(float f) {
  unsigned int u = __float_as_uint(f);
  unsigned int r = (u + 0x7fffu + ((u >> 16) & 1u)) >> 16;
  return (unsigned short)r;
}

// Low 32 bits of a generic pointer to __shared__ == AS(3) LDS byte offset.
static __device__ inline unsigned lds_off(const void* p) {
  return (unsigned)(unsigned long long)p;
}

// A-operand (M=16 x K=32): lane holds row m=lane%16; K elems at
// offsets 8*(lane/16)..+7 and 16+8*(lane/16)..+7 (ISA 7.12.2, 16-bit A).
static __device__ inline v16bf load_a_frag(const unsigned short* p, int stride) {
  int lane = threadIdx.x & 31;
  int m = lane & 15, half = lane >> 4;
  const unsigned short* row = p + (size_t)m * stride + 8 * half;
  FragBf16 f;
  f.u[0] = *(const uint4*)(row);
  f.u[1] = *(const uint4*)(row + 16);
  return f.v;
}

// B-operand (K=32 x N=16) from Bt[n][k] row-major: lane holds col n=lane%16,
// K elems at offsets 8*(lane/16)..+7 and 16+8*(lane/16)..+7 (matches A).
static __device__ inline v16bf load_b_frag(const unsigned short* p, int stride) {
  int lane = threadIdx.x & 31;
  int n = lane & 15, half = lane >> 4;
  const unsigned short* row = p + (size_t)n * stride + 8 * half;
  FragBf16 f;
  f.u[0] = *(const uint4*)(row);
  f.u[1] = *(const uint4*)(row + 16);
  return f.v;
}

static __device__ inline v8f wmma_bf16(v16bf a, v16bf b, v8f c) {
  return __builtin_amdgcn_wmma_f32_16x16x32_bf16(false, a, false, b, (short)0, c,
                                                 false, false);
}

// Per-lane async global->LDS copy of 32 bytes (two B128 chunks; the inst
// offset applies to BOTH the LDS and global address, ISA 10.2/15.18.3).
static __device__ inline void async_copy_32B(unsigned lds_byte,
                                             const unsigned short* g) {
  asm volatile("global_load_async_to_lds_b128 %0, %1, off"
               :: "v"(lds_byte), "v"(g) : "memory");
  asm volatile("global_load_async_to_lds_b128 %0, %1, off offset:16"
               :: "v"(lds_byte), "v"(g) : "memory");
}

static __device__ inline void wait_asynccnt0() {
  asm volatile("s_wait_asynccnt 0x0" ::: "memory");
}

// TDM 2-D tile load: tile_h rows x tile_w bf16 elems, row stride (elems),
// contiguous into LDS. D# layout per CDNA5 ISA.
static __device__ inline void tdm_load_tile_2d(unsigned lds_byte,
                                               const void* gaddr,
                                               unsigned tile_w, unsigned tile_h,
                                               unsigned stride_elems) {
#if __has_builtin(__builtin_amdgcn_tensor_load_to_lds)
  unsigned long long ga = (unsigned long long)gaddr;
  v4u g0;
  g0[0] = 1u;                                   // count=1, is_restore=0
  g0[1] = lds_byte;                             // lds_addr
  g0[2] = (unsigned)(ga & 0xffffffffu);         // global_addr[31:0]
  g0[3] = (unsigned)((ga >> 32) & 0x01ffffffu)  // global_addr[56:32]
          | (2u << 30);                         // type=2 ("image")
  v8i g1;
  g1[0] = (int)(1u << 16);                      // data_size=1 -> 2 bytes
  g1[1] = (int)((tile_w & 0xffffu) << 16);      // tensor_dim0[15:0]
  g1[2] = (int)(((tile_w >> 16) & 0xffffu) | ((tile_h & 0xffffu) << 16));
  g1[3] = (int)(((tile_h >> 16) & 0xffffu) | ((tile_w & 0xffffu) << 16)); // tile_dim0
  g1[4] = (int)(tile_h & 0xffffu);              // tile_dim1 (tile_dim2=0)
  g1[5] = (int)stride_elems;                    // tensor_dim0_stride[31:0]
  g1[6] = 0;
  g1[7] = 0;
  v4i gz = {0, 0, 0, 0};
#if __has_include(<hip/amd_detail/amd_gfx1250_TDM.h>)
  v8i gz8 = {0, 0, 0, 0, 0, 0, 0, 0};
  __builtin_amdgcn_tensor_load_to_lds(g0, g1, gz, gz, gz8, 0);
#else
  __builtin_amdgcn_tensor_load_to_lds(g0, g1, gz, gz, 0);
#endif
  __builtin_amdgcn_s_wait_tensorcnt(0);
#else
  (void)lds_byte; (void)gaddr; (void)tile_w; (void)tile_h; (void)stride_elems;
#endif
}

__global__ void cvt_f32_bf16(const float* __restrict__ in,
                             unsigned short* __restrict__ out, long long n) {
  long long i = (long long)blockIdx.x * blockDim.x + threadIdx.x;
  if (i < n) out[i] = f2bf(in[i]);
}

// in [K][N] f32 -> out [N][K] bf16
__global__ void cvt_transpose_bf16(const float* __restrict__ in,
                                   unsigned short* __restrict__ out, int K, int N) {
  long long i = (long long)blockIdx.x * blockDim.x + threadIdx.x;
  if (i < (long long)K * N) {
    int n = (int)(i / K), k = (int)(i % K);
    out[i] = f2bf(in[(long long)k * N + n]);
  }
}

// Vb [B*S][NKV*HD] bf16 -> Vt [B][NKV][HD][S] bf16
__global__ void transpose_v(const unsigned short* __restrict__ Vb,
                            unsigned short* __restrict__ Vt) {
  long long i = (long long)blockIdx.x * blockDim.x + threadIdx.x;
  if (i < (long long)B_ * S_ * (NKV_ * HD_)) {
    int n = (int)(i & 511);
    long long m = i >> 9;
    int b = (int)(m >> 11), s = (int)(m & 2047);
    int kv = n >> 7, d = n & 127;
    Vt[((((size_t)b * NKV_ + kv) * HD_ + d) << 11) + s] = Vb[i];
  }
}

// C[M,N] = alpha * A[M,K] @ Bt[N,K]^T, bf16 in, f32 accumulate.
// Workgroup tile 128x64, 8 waves in 4x2, each wave 32x32 (2x2 WMMA tiles).
// A tile staged via per-lane GLOBAL_LOAD_ASYNC_TO_LDS_B128 (ASYNCcnt),
// B tile staged via one TENSOR_LOAD_TO_LDS DMA from wave 0 (TENSORcnt).
__global__ __launch_bounds__(256) void gemm_bf16(
    const unsigned short* __restrict__ A, const unsigned short* __restrict__ Bt,
    float* __restrict__ Cf, unsigned short* __restrict__ Cb,
    int M, int N, int K, float alpha) {
  __shared__ unsigned short sA[128 * 32];
  __shared__ unsigned short sB[64 * 32];
  int tid = threadIdx.x;
  int wid = tid >> 5, lane = tid & 31;
  int bm = blockIdx.y * 128, bn = blockIdx.x * 64;
  int wm = (wid >> 1) * 32, wn = (wid & 1) * 32;
  const v8f vzero = {0.f, 0.f, 0.f, 0.f, 0.f, 0.f, 0.f, 0.f};
  v8f acc[2][2];
  acc[0][0] = vzero; acc[0][1] = vzero; acc[1][0] = vzero; acc[1][1] = vzero;

  int ar = tid >> 1, ac = (tid & 1) * 16;              // A: 32B per thread
  unsigned aLds = lds_off(sA + ar * 32 + ac);
  unsigned bLds = lds_off(sB);

  for (int k0 = 0; k0 < K; k0 += 32) {
    async_copy_32B(aLds, A + (size_t)(bm + ar) * K + k0 + ac);
    if (wid == 0) {
      tdm_load_tile_2d(bLds, Bt + (size_t)bn * K + k0, 32u, 64u, (unsigned)K);
    }
    wait_asynccnt0();
    __syncthreads();

    v16bf a0 = load_a_frag(sA + (wm + 0) * 32, 32);
    v16bf a1 = load_a_frag(sA + (wm + 16) * 32, 32);
    v16bf b0 = load_b_frag(sB + (wn + 0) * 32, 32);
    v16bf b1 = load_b_frag(sB + (wn + 16) * 32, 32);
    acc[0][0] = wmma_bf16(a0, b0, acc[0][0]);
    acc[0][1] = wmma_bf16(a0, b1, acc[0][1]);
    acc[1][0] = wmma_bf16(a1, b0, acc[1][0]);
    acc[1][1] = wmma_bf16(a1, b1, acc[1][1]);
    __syncthreads();
  }
  int half = lane >> 4, col = lane & 15;
  for (int mi = 0; mi < 2; ++mi)
    for (int ni = 0; ni < 2; ++ni)
      for (int r = 0; r < 8; ++r) {
        int row = bm + wm + mi * 16 + r + 8 * half;
        int c = bn + wn + ni * 16 + col;
        float v = acc[mi][ni][r] * alpha;
        if (Cf) Cf[(size_t)row * N + c] = v;
        if (Cb) Cb[(size_t)row * N + c] = f2bf(v);
      }
}

// One wave per 16 query rows; 8 waves/block -> 128 q rows per block.
// Pass 1: streaming online max/sumexp over all 2048 keys (registers only).
// Pass 2: recompute scores, write normalized weights (f32, an output),
// re-layout P via LDS, accumulate O = P @ V^T with WMMA.
__global__ __launch_bounds__(256) void gqa_attention(
    const unsigned short* __restrict__ Qb,   // [B*S][H] bf16, pre-scaled
    const unsigned short* __restrict__ Kb,   // [B*S][NKV*HD] bf16
    const unsigned short* __restrict__ Vt,   // [B][NKV][HD][S] bf16
    const float* __restrict__ bias,          // [B][1][S][S] f32
    float* __restrict__ Wout,                // [B][NH][S][S] f32
    unsigned short* __restrict__ ctx) {      // [B*S][H] bf16
  __shared__ unsigned short sP[8][16 * 32];
  int wid = threadIdx.x >> 5, lane = threadIdx.x & 31;
  int half = lane >> 4, col = lane & 15;
  int b = blockIdx.z, h = blockIdx.y;
  int q0 = blockIdx.x * 128 + wid * 16;
  int kv = h >> 2;  // N_REP = 4

  const unsigned short* Qw = Qb + (size_t)(b * S_ + q0) * H_ + h * HD_;
  const unsigned short* Kw = Kb + (size_t)b * S_ * (NKV_ * HD_) + kv * HD_;
  const unsigned short* Vw = Vt + (size_t)(b * NKV_ + kv) * HD_ * S_;
  const float* Bw = bias + ((size_t)b * S_ + q0) * S_;
  float* Ww = Wout + (((size_t)(b * NH_ + h)) * S_ + q0) * S_;

  const v8f vzero = {0.f, 0.f, 0.f, 0.f, 0.f, 0.f, 0.f, 0.f};

  v16bf qf[4];
  for (int f = 0; f < 4; ++f) qf[f] = load_a_frag(Qw + f * 32, H_);

  v8f m, lsum;
  for (int r = 0; r < 8; ++r) { m[r] = -3.0e38f; lsum[r] = 0.f; }

  // ---- pass 1: max & sumexp ----
  for (int kt = 0; kt < S_ / 16; ++kt) {
    int k0 = kt * 16;
    v8f s = vzero;
    for (int f = 0; f < 4; ++f) {
      v16bf bf = load_b_frag(Kw + (size_t)k0 * (NKV_ * HD_) + f * 32, NKV_ * HD_);
      s = wmma_bf16(qf[f], bf, s);
    }
    for (int r = 0; r < 8; ++r)
      s[r] += Bw[(size_t)(r + 8 * half) * S_ + k0 + col];
    for (int r = 0; r < 8; ++r) {
      float tmax = s[r];
      for (int d = 1; d < 16; d <<= 1) tmax = fmaxf(tmax, __shfl_xor(tmax, d, 16));
      float mn = fmaxf(m[r], tmax);
      float p = __expf(s[r] - mn);
      for (int d = 1; d < 16; d <<= 1) p += __shfl_xor(p, d, 16);
      lsum[r] = lsum[r] * __expf(m[r] - mn) + p;
      m[r] = mn;
    }
  }
  v8f rinv;
  for (int r = 0; r < 8; ++r) rinv[r] = 1.0f / lsum[r];

  // ---- pass 2: weights out + O accumulate ----
  v8f o[8];
  for (int db = 0; db < 8; ++db) o[db] = vzero;
  unsigned short* sPw = sP[wid];

  for (int k0 = 0; k0 < S_; k0 += 32) {
    for (int kk = 0; kk < 2; ++kk) {
      int kc = k0 + kk * 16;
      v8f s = vzero;
      for (int f = 0; f < 4; ++f) {
        v16bf bf = load_b_frag(Kw + (size_t)kc * (NKV_ * HD_) + f * 32, NKV_ * HD_);
        s = wmma_bf16(qf[f], bf, s);
      }
      for (int r = 0; r < 8; ++r) {
        float sc = s[r] + Bw[(size_t)(r + 8 * half) * S_ + kc + col];
        float w = __expf(sc - m[r]) * rinv[r];
        Ww[(size_t)(r + 8 * half) * S_ + kc + col] = w;
        sPw[(r + 8 * half) * 32 + kk * 16 + col] = f2bf(w);
      }
    }
    v16bf pA = load_a_frag(sPw, 32);
    for (int db = 0; db < 8; ++db) {
      v16bf bv = load_b_frag(Vw + (size_t)db * 16 * S_ + k0, S_);
      o[db] = wmma_bf16(pA, bv, o[db]);
    }
  }

  unsigned short* cw = ctx + (size_t)(b * S_ + q0) * H_ + h * HD_;
  for (int db = 0; db < 8; ++db)
    for (int r = 0; r < 8; ++r)
      cw[(size_t)(r + 8 * half) * H_ + db * 16 + col] = f2bf(o[db][r]);
}

extern "C" void kernel_launch(void* const* d_in, const int* in_sizes, int n_in,
                              void* d_out, int out_size, void* d_ws, size_t ws_size,
                              hipStream_t stream) {
  (void)in_sizes; (void)n_in; (void)out_size; (void)ws_size;
  const float* x    = (const float*)d_in[0];
  const float* bias = (const float*)d_in[1];
  const float* Wq   = (const float*)d_in[2];
  const float* Wk   = (const float*)d_in[3];
  const float* Wv   = (const float*)d_in[4];
  const float* Wo   = (const float*)d_in[5];
  float* out  = (float*)d_out;                       // [B,S,H]
  float* wout = out + (size_t)B_ * S_ * H_;          // [B,NH,S,S]

  char* w = (char*)d_ws;
  unsigned short* xb  = (unsigned short*)(w);                       // 16 MB
  unsigned short* Wqt = (unsigned short*)(w + (16u << 20));         //  8 MB
  unsigned short* Wkt = (unsigned short*)(w + (24u << 20));         //  2 MB
  unsigned short* Wvt = (unsigned short*)(w + (26u << 20));         //  2 MB
  unsigned short* Wot = (unsigned short*)(w + (28u << 20));         //  8 MB
  unsigned short* Qb  = (unsigned short*)(w + (36u << 20));         // 16 MB
  unsigned short* Kb  = (unsigned short*)(w + (52u << 20));         //  4 MB
  unsigned short* Vb  = (unsigned short*)(w + (56u << 20));         //  4 MB
  unsigned short* Vt  = (unsigned short*)(w + (60u << 20));         //  4 MB
  unsigned short* ctx = (unsigned short*)(w + (64u << 20));         // 16 MB

  const long long nx = (long long)B_ * S_ * H_;  // 8,388,608
  cvt_f32_bf16<<<(unsigned)((nx + 255) / 256), 256, 0, stream>>>(x, xb, nx);
  cvt_transpose_bf16<<<(H_ * H_) / 256, 256, 0, stream>>>(Wq, Wqt, H_, H_);
  cvt_transpose_bf16<<<(H_ * NKV_ * HD_) / 256, 256, 0, stream>>>(Wk, Wkt, H_, NKV_ * HD_);
  cvt_transpose_bf16<<<(H_ * NKV_ * HD_) / 256, 256, 0, stream>>>(Wv, Wvt, H_, NKV_ * HD_);
  cvt_transpose_bf16<<<(H_ * H_) / 256, 256, 0, stream>>>(Wo, Wot, H_, H_);

  int Mrows = B_ * S_;  // 4096
  gemm_bf16<<<dim3(H_ / 64, Mrows / 128), 256, 0, stream>>>(
      xb, Wqt, nullptr, Qb, Mrows, H_, H_, SCALE_);
  gemm_bf16<<<dim3((NKV_ * HD_) / 64, Mrows / 128), 256, 0, stream>>>(
      xb, Wkt, nullptr, Kb, Mrows, NKV_ * HD_, H_, 1.0f);
  gemm_bf16<<<dim3((NKV_ * HD_) / 64, Mrows / 128), 256, 0, stream>>>(
      xb, Wvt, nullptr, Vb, Mrows, NKV_ * HD_, H_, 1.0f);

  transpose_v<<<(unsigned)(((long long)Mrows * NKV_ * HD_) / 256), 256, 0, stream>>>(Vb, Vt);

  gqa_attention<<<dim3(S_ / 128, NH_, B_), 256, 0, stream>>>(
      Qb, Kb, Vt, bias, wout, ctx);

  gemm_bf16<<<dim3(H_ / 64, Mrows / 128), 256, 0, stream>>>(
      ctx, Wot, out, nullptr, Mrows, H_, H_, 1.0f);
}
